// DeformableAttSegmHead_34196529610802
// MI455X (gfx1250) — matrix operand order repaired and checked by reading, not compiled, not run on hardware
//
#include <hip/hip_runtime.h>
#include <cstdint>
#include <cstddef>

// ---------------- problem constants (from reference) ----------------
#define TT   6
#define QQ   300
#define DD   256
#define NHH  8
#define DHH  32
#define LLV  4
#define PPT  7          // P = PC + (T-1)*PT = 2 + 5
#define PCC  2
#define TQ   (TT*QQ)    // 1800
#define DFFN 1024
#define NVAL 130560     // T * (128^2+64^2+32^2+16^2)

typedef __attribute__((ext_vector_type(16))) __bf16 v16bf;
typedef __attribute__((ext_vector_type(8)))  __bf16 v8bf;
typedef __attribute__((ext_vector_type(4)))  __bf16 v4bf;
typedef __attribute__((ext_vector_type(8)))  float  v8f;
typedef __attribute__((ext_vector_type(4)))  float  v4f;

__device__ __forceinline__ __bf16 f2bf(float f) {
  unsigned u = __builtin_bit_cast(unsigned, f);
  u += 0x7FFFu + ((u >> 16) & 1u);              // round-to-nearest-even
  unsigned short h = (unsigned short)(u >> 16);
  return __builtin_bit_cast(__bf16, h);
}

__device__ __forceinline__ float wave_sum(float v) {
#pragma unroll
  for (int o = 16; o > 0; o >>= 1) v += __shfl_xor(v, o, 32);
  return v;
}

// ---------------- f32 -> bf16 bulk conversion (n multiple of 4) ----------------
__global__ __launch_bounds__(256) void cvt_bf16(const float* __restrict__ a,
                                                __bf16* __restrict__ o, int n) {
  int i = (blockIdx.x * blockDim.x + threadIdx.x) * 4;
  if (i < n) {
    v4f x = *(const v4f*)(a + i);
    v4bf y;
#pragma unroll
    for (int j = 0; j < 4; j++) y[j] = f2bf(x[j]);
    *(v4bf*)(o + i) = y;
  }
}

// qkin_bf = bf16(tgt_masks + query_pos)
__global__ __launch_bounds__(256) void add_cvt_bf16(const float* __restrict__ a,
                                                    const float* __restrict__ b,
                                                    __bf16* __restrict__ o, int n) {
  int i = (blockIdx.x * blockDim.x + threadIdx.x) * 4;
  if (i < n) {
    v4f x = *(const v4f*)(a + i);
    v4f y = *(const v4f*)(b + i);
    v4bf z;
#pragma unroll
    for (int j = 0; j < 4; j++) z[j] = f2bf(x[j] + y[j]);
    *(v4bf*)(o + i) = z;
  }
}

// ---------------- WMMA GEMM: C[M,N] = A[M,K] @ W[N,K]^T + bias ----------------
// bf16 operands, f32 accumulate. One wave owns a 32x32 tile (2x2 WMMA blocking):
// 6 fragment loads feed 4 v_wmma per K-step, zero conversion VALU in the loop.
// grid.x covers N (8 waves * 32 cols = 256 cols/block), grid.y covers M (32 rows)
// with N innermost so the bf16 weights stay resident in L2 and A streams once.
__global__ __launch_bounds__(256) void gemm_bf16_wmma(
    const __bf16* __restrict__ A, const __bf16* __restrict__ W,
    const float* __restrict__ bias, float* __restrict__ Cf,
    __bf16* __restrict__ Cb, int M, int N, int K, int relu)
{
  const int lane = threadIdx.x & 31;
  const int wv   = threadIdx.x >> 5;
  const int mt   = blockIdx.y;                 // 32-row tile
  const int nt   = blockIdx.x * 8 + wv;        // 32-col tile
  if (nt * 32 >= N) return;

  const int l15 = lane & 15;
  int r0 = mt * 32 + l15;       if (r0 > M - 1) r0 = M - 1;
  int r1 = mt * 32 + 16 + l15;  if (r1 > M - 1) r1 = M - 1;
  const int c0 = nt * 32 + l15;
  const int c1 = c0 + 16;
  const int hiA = (lane >> 4) << 4;   // 0 or 16 halves
  const int hiB = (lane >> 4) << 4;   // 0 or 16 halves

  const __bf16* a0p = A + (size_t)r0 * K + hiA;
  const __bf16* a1p = A + (size_t)r1 * K + hiA;
  const __bf16* b0p = W + (size_t)c0 * K + hiB;
  const __bf16* b1p = W + (size_t)c1 * K + hiB;

  v8f acc00 = {0.f,0.f,0.f,0.f,0.f,0.f,0.f,0.f};
  v8f acc01 = acc00, acc10 = acc00, acc11 = acc00;

  for (int kb = 0; kb < K; kb += 32) {
    __builtin_prefetch(a0p + kb + 128, 0, 1);
    __builtin_prefetch(a1p + kb + 128, 0, 1);
    v8bf a0lo = *(const v8bf*)(a0p + kb);
    v8bf a0hi = *(const v8bf*)(a0p + kb + 8);
    v8bf a1lo = *(const v8bf*)(a1p + kb);
    v8bf a1hi = *(const v8bf*)(a1p + kb + 8);
    v16bf b0  = *(const v16bf*)(b0p + kb);
    v16bf b1  = *(const v16bf*)(b1p + kb);
    v16bf a0  = __builtin_shufflevector(a0lo, a0hi, 0,1,2,3,4,5,6,7,8,9,10,11,12,13,14,15);
    v16bf a1  = __builtin_shufflevector(a1lo, a1hi, 0,1,2,3,4,5,6,7,8,9,10,11,12,13,14,15);
    acc00 = __builtin_amdgcn_wmma_f32_16x16x32_bf16(false, a0, false, b0, (short)0, acc00, false, false);
    acc01 = __builtin_amdgcn_wmma_f32_16x16x32_bf16(false, a0, false, b1, (short)0, acc01, false, false);
    acc10 = __builtin_amdgcn_wmma_f32_16x16x32_bf16(false, a1, false, b0, (short)0, acc10, false, false);
    acc11 = __builtin_amdgcn_wmma_f32_16x16x32_bf16(false, a1, false, b1, (short)0, acc11, false, false);
  }

  const float bv0 = bias ? bias[c0] : 0.0f;
  const float bv1 = bias ? bias[c1] : 0.0f;
  const int rb = (lane >> 4) << 3;
#pragma unroll
  for (int v = 0; v < 8; v++) {
    int ra = mt * 32 + rb + v;
    int rc = ra + 16;
    if (ra < M) {
      float o0 = acc00[v] + bv0, o1 = acc01[v] + bv1;
      if (relu) { o0 = fmaxf(o0, 0.f); o1 = fmaxf(o1, 0.f); }
      if (Cb) { Cb[(size_t)ra * N + c0] = f2bf(o0); Cb[(size_t)ra * N + c1] = f2bf(o1); }
      else    { Cf[(size_t)ra * N + c0] = o0;       Cf[(size_t)ra * N + c1] = o1; }
    }
    if (rc < M) {
      float o0 = acc10[v] + bv0, o1 = acc11[v] + bv1;
      if (relu) { o0 = fmaxf(o0, 0.f); o1 = fmaxf(o1, 0.f); }
      if (Cb) { Cb[(size_t)rc * N + c0] = f2bf(o0); Cb[(size_t)rc * N + c1] = f2bf(o1); }
      else    { Cf[(size_t)rc * N + c0] = o0;       Cf[(size_t)rc * N + c1] = o1; }
    }
  }
}

// ---------------- self-attention over frames: one wave per (q,h) ----------------
__global__ __launch_bounds__(128) void selfattn(
    const float* __restrict__ qh, const float* __restrict__ kh,
    const float* __restrict__ vh, __bf16* __restrict__ sa)
{
  int lane = threadIdx.x & 31;
  int wv   = threadIdx.x >> 5;
  int idx  = blockIdx.x * 4 + wv;
  if (idx >= QQ * NHH) return;
  int q = idx / NHH, h = idx % NHH;
  size_t base = (size_t)q * DD + h * DHH + lane;

  float qv[TT], kv[TT], vv[TT];
#pragma unroll
  for (int i = 0; i < TT; i++) {
    qv[i] = qh[(size_t)i * QQ * DD + base];
    kv[i] = kh[(size_t)i * QQ * DD + base];
    vv[i] = vh[(size_t)i * QQ * DD + base];
  }
  float lg[TT][TT];
#pragma unroll
  for (int i = 0; i < TT; i++)
#pragma unroll
    for (int j = 0; j < TT; j++)
      lg[i][j] = wave_sum(qv[i] * kv[j]) * 0.17677669529663687f; // 1/sqrt(32)

#pragma unroll
  for (int i = 0; i < TT; i++) {
    float m = lg[i][0];
#pragma unroll
    for (int j = 1; j < TT; j++) m = fmaxf(m, lg[i][j]);
    float s = 0.f, o = 0.f, e[TT];
#pragma unroll
    for (int j = 0; j < TT; j++) { e[j] = __expf(lg[i][j] - m); s += e[j]; }
    float inv = 1.0f / s;
#pragma unroll
    for (int j = 0; j < TT; j++) o += e[j] * inv * vv[j];
    sa[(size_t)i * QQ * DD + base] = f2bf(o);
  }
}

// ---------------- residual + layernorm (D=256) ----------------
// outf = LN(x+res); outbf = bf16(outf + qpos) if qpos else bf16(outf)
__global__ __launch_bounds__(256) void add_ln(
    const float* __restrict__ x, const float* __restrict__ res,
    const float* __restrict__ g, const float* __restrict__ b,
    float* __restrict__ outf, __bf16* __restrict__ outbf,
    const float* __restrict__ qpos)
{
  int row = blockIdx.x, tid = threadIdx.x;
  size_t o = (size_t)row * DD + tid;
  float v = x[o] + res[o];
  __shared__ float red[8];
  float s = wave_sum(v);
  if ((tid & 31) == 0) red[tid >> 5] = s;
  __syncthreads();
  float tot = 0.f;
#pragma unroll
  for (int i = 0; i < 8; i++) tot += red[i];
  float mu = tot * (1.0f / DD);
  __syncthreads();
  float d = v - mu;
  float s2 = wave_sum(d * d);
  if ((tid & 31) == 0) red[tid >> 5] = s2;
  __syncthreads();
  float tv = 0.f;
#pragma unroll
  for (int i = 0; i < 8; i++) tv += red[i];
  float y = d * rsqrtf(tv * (1.0f / DD) + 1e-5f) * g[tid] + b[tid];
  outf[o] = y;
  if (outbf) outbf[o] = f2bf(qpos ? (y + qpos[o]) : y);
}

// ---------------- softmax over 28 sampling weights per (n,h), in place ----------------
__global__ void att_softmax(float* __restrict__ att) {
  int i = blockIdx.x * blockDim.x + threadIdx.x;
  if (i >= TQ * NHH) return;
  float* p = att + (size_t)(i / NHH) * (NHH * LLV * PPT) + (i % NHH) * (LLV * PPT);
  float m = p[0];
#pragma unroll
  for (int j = 1; j < LLV * PPT; j++) m = fmaxf(m, p[j]);
  float s = 0.f;
#pragma unroll
  for (int j = 0; j < LLV * PPT; j++) { float e = __expf(p[j] - m); p[j] = e; s += e; }
  float inv = 1.0f / s;
#pragma unroll
  for (int j = 0; j < LLV * PPT; j++) p[j] *= inv;
}

// ---------------- deformable bilinear sampling: one wave per (n,h), lane = channel ----------------
__global__ __launch_bounds__(256) void deform_sample(
    const float* __restrict__ value, const float* __restrict__ off,
    const float* __restrict__ att,   const float* __restrict__ refp,
    const float* __restrict__ vr,    __bf16* __restrict__ out)
{
  const int n    = blockIdx.x;          // 0..TQ-1
  const int h    = threadIdx.x >> 5;    // 0..7
  const int lane = threadIdx.x & 31;    // channel in head
  const int t = n / QQ, q = n % QQ;

  const int HL[LLV]     = {128, 64, 32, 16};
  const int LSTART[LLV] = {0, 98304, 122880, 129024};

  float acc = 0.f;
#pragma unroll
  for (int l = 0; l < LLV; l++) {
    const int Hl = HL[l], Wl = HL[l];
    const int ls = LSTART[l];
    const float vrx = vr[l * 2 + 0], vry = vr[l * 2 + 1];
    const float invW = 1.0f / (float)Wl, invH = 1.0f / (float)Hl;
#pragma unroll
    for (int p = 0; p < PPT; p++) {
      int fr   = (p < PCC) ? t : (((p - PCC) < t) ? (p - PCC) : (p - PCC + 1));
      int ridx = (p < PCC) ? n : (fr * QQ + q);
      float rx = refp[ridx * 2 + 0] * vrx;
      float ry = refp[ridx * 2 + 1] * vry;
      size_t ob = (size_t)n * (NHH * LLV * PPT * 2) + (size_t)(((h * LLV + l) * PPT + p) * 2);
      float lx = rx + off[ob]     * invW;
      float ly = ry + off[ob + 1] * invH;
      float aw = att[(size_t)n * (NHH * LLV * PPT) + (h * LLV * PPT + l * PPT + p)];
      float x = lx * (float)Wl - 0.5f, y = ly * (float)Hl - 0.5f;
      float xf = floorf(x), yf = floorf(y);
      float fx = x - xf, fy = y - yf;
      int x0 = (int)xf, y0 = (int)yf;
      int fb = ls + fr * Hl * Wl;
      float sample = 0.f;
#pragma unroll
      for (int dy = 0; dy < 2; dy++) {
#pragma unroll
        for (int dx = 0; dx < 2; dx++) {
          int xi = x0 + dx, yi = y0 + dy;
          float w = (dy ? fy : 1.f - fy) * (dx ? fx : 1.f - fx);
          if (xi >= 0 && xi < Wl && yi >= 0 && yi < Hl) {
            size_t vrow = (size_t)(fb + yi * Wl + xi);
            sample += w * value[vrow * DD + h * DHH + lane];
          }
        }
      }
      acc += aw * sample;
    }
  }
  out[(size_t)n * DD + h * DHH + lane] = f2bf(acc);
}

// ---------------- host ----------------
extern "C" void kernel_launch(void* const* d_in, const int* in_sizes, int n_in,
                              void* d_out, int out_size, void* d_ws, size_t ws_size,
                              hipStream_t stream) {
  const float* src  = (const float*)d_in[0];
  const float* tgtm = (const float*)d_in[1];
  const float* qpos = (const float*)d_in[2];
  const float* refp = (const float*)d_in[3];
  const float* vr   = (const float*)d_in[4];
  const float* Wq   = (const float*)d_in[7];  const float* bq   = (const float*)d_in[8];
  const float* Wk   = (const float*)d_in[9];  const float* bk   = (const float*)d_in[10];
  const float* Wv   = (const float*)d_in[11]; const float* bv   = (const float*)d_in[12];
  const float* Wo   = (const float*)d_in[13]; const float* bo   = (const float*)d_in[14];
  const float* n2g  = (const float*)d_in[15]; const float* n2b  = (const float*)d_in[16];
  const float* Wval = (const float*)d_in[17]; const float* bval = (const float*)d_in[18];
  const float* Woff = (const float*)d_in[19]; const float* boff = (const float*)d_in[20];
  const float* Watt = (const float*)d_in[21]; const float* batt = (const float*)d_in[22];
  const float* Wout = (const float*)d_in[23]; const float* bout = (const float*)d_in[24];
  const float* n1g  = (const float*)d_in[25]; const float* n1b  = (const float*)d_in[26];
  const float* W1   = (const float*)d_in[27]; const float* b1   = (const float*)d_in[28];
  const float* W2   = (const float*)d_in[29]; const float* b2   = (const float*)d_in[30];
  const float* n3g  = (const float*)d_in[31]; const float* n3b  = (const float*)d_in[32];
  float* outp = (float*)d_out;

  float* ws = (float*)d_ws;
  size_t wo = 0;
  auto allocF = [&](size_t nf) { float* p = ws + wo; wo += nf; return p; };
  auto allocH = [&](size_t nh) { __bf16* p = (__bf16*)(ws + wo); wo += (nh + 1) / 2; return p; };

  // f32 scratch
  float* qh    = allocF((size_t)TQ * DD);
  float* kh    = allocF((size_t)TQ * DD);
  float* vhb   = allocF((size_t)TQ * DD);
  float* sao   = allocF((size_t)TQ * DD);
  float* tgt1  = allocF((size_t)TQ * DD);
  float* value = allocF((size_t)NVAL * DD);
  float* offb  = allocF((size_t)TQ * NHH * LLV * PPT * 2);
  float* attl  = allocF((size_t)TQ * NHH * LLV * PPT);
  float* cross = allocF((size_t)TQ * DD);
  float* tgt2  = allocF((size_t)TQ * DD);
  float* ffnt  = allocF((size_t)TQ * DD);
  // bf16 scratch (GEMM operands)
  __bf16* qkin_b = allocH((size_t)TQ * DD);
  __bf16* tgtm_b = allocH((size_t)TQ * DD);
  __bf16* sa_b   = allocH((size_t)TQ * DD);
  __bf16* qry_b  = allocH((size_t)TQ * DD);
  __bf16* dco_b  = allocH((size_t)TQ * DD);
  __bf16* tgt2_b = allocH((size_t)TQ * DD);
  __bf16* ffnh_b = allocH((size_t)TQ * DFFN);
  __bf16* src_b  = allocH((size_t)NVAL * DD);
  __bf16* Wq_b   = allocH((size_t)DD * DD);
  __bf16* Wk_b   = allocH((size_t)DD * DD);
  __bf16* Wv_b   = allocH((size_t)DD * DD);
  __bf16* Wo_b   = allocH((size_t)DD * DD);
  __bf16* Wval_b = allocH((size_t)DD * DD);
  __bf16* Woff_b = allocH((size_t)(NHH * LLV * PPT * 2) * DD);
  __bf16* Watt_b = allocH((size_t)(NHH * LLV * PPT) * DD);
  __bf16* Wout_b = allocH((size_t)DD * DD);
  __bf16* W1_b   = allocH((size_t)DFFN * DD);
  __bf16* W2_b   = allocH((size_t)DD * DFFN);
  (void)ws_size; (void)in_sizes; (void)n_in; (void)out_size;

  auto cvt = [&](const float* a, __bf16* o, int n) {
    cvt_bf16<<<(n / 4 + 255) / 256, 256, 0, stream>>>(a, o, n);
  };
  auto gemm = [&](const __bf16* A, const __bf16* W, const float* B,
                  float* Cf, __bf16* Cb, int M, int N, int K, int relu) {
    dim3 grid((N + 255) / 256, (M + 31) / 32);
    gemm_bf16_wmma<<<grid, dim3(256), 0, stream>>>(A, W, B, Cf, Cb, M, N, K, relu);
  };

  // 0) operand conversions (weights are tiny; src is the one big stream)
  cvt(Wq, Wq_b, DD * DD);       cvt(Wk, Wk_b, DD * DD);
  cvt(Wv, Wv_b, DD * DD);       cvt(Wo, Wo_b, DD * DD);
  cvt(Wval, Wval_b, DD * DD);   cvt(Wout, Wout_b, DD * DD);
  cvt(Woff, Woff_b, NHH * LLV * PPT * 2 * DD);
  cvt(Watt, Watt_b, NHH * LLV * PPT * DD);
  cvt(W1, W1_b, DFFN * DD);     cvt(W2, W2_b, DD * DFFN);
  cvt(tgtm, tgtm_b, TQ * DD);
  cvt(src, src_b, NVAL * DD);
  // 1) qk_in = tgt_masks + query_pos (bf16)
  add_cvt_bf16<<<(TQ * DD / 4 + 255) / 256, 256, 0, stream>>>(tgtm, qpos, qkin_b, TQ * DD);
  // 2-4) projections
  gemm(qkin_b, Wq_b, bq, qh,  nullptr, TQ, DD, DD, 0);
  gemm(qkin_b, Wk_b, bk, kh,  nullptr, TQ, DD, DD, 0);
  gemm(tgtm_b, Wv_b, bv, vhb, nullptr, TQ, DD, DD, 0);
  // 5) frame self-attention (emits bf16 for the O-projection)
  selfattn<<<(QQ * NHH + 3) / 4, 128, 0, stream>>>(qh, kh, vhb, sa_b);
  // 6) output projection
  gemm(sa_b, Wo_b, bo, sao, nullptr, TQ, DD, DD, 0);
  // 7) tgt = LN(tgt_masks + sa); query_bf = bf16(tgt + query_pos)
  add_ln<<<TQ, DD, 0, stream>>>(sao, tgtm, n2g, n2b, tgt1, qry_b, qpos);
  // 8) value projection (the big GEMM, f32 out for accurate sampling)
  gemm(src_b, Wval_b, bval, value, nullptr, NVAL, DD, DD, 0);
  // 9-10) offsets & attention logits
  gemm(qry_b, Woff_b, boff, offb, nullptr, TQ, NHH * LLV * PPT * 2, DD, 0);
  gemm(qry_b, Watt_b, batt, attl, nullptr, TQ, NHH * LLV * PPT, DD, 0);
  // 11) softmax over L*P per (n,h)
  att_softmax<<<(TQ * NHH + 255) / 256, 256, 0, stream>>>(attl);
  // 12) deformable sampling (emits bf16 for the out-projection)
  deform_sample<<<TQ, 256, 0, stream>>>(value, offb, attl, refp, vr, dco_b);
  // 13) cross projection
  gemm(dco_b, Wout_b, bout, cross, nullptr, TQ, DD, DD, 0);
  // 14) tgt = LN(tgt + cross) ; tgt2_bf for FFN input
  add_ln<<<TQ, DD, 0, stream>>>(cross, tgt1, n1g, n1b, tgt2, tgt2_b, nullptr);
  // 15-16) FFN (hidden kept in bf16: GEMM epilogue converts)
  gemm(tgt2_b, W1_b, b1, nullptr, ffnh_b, TQ, DFFN, DD, 1);
  gemm(ffnh_b, W2_b, b2, ffnt, nullptr, TQ, DD, DFFN, 0);
  // 17) final LN -> output
  add_ln<<<TQ, DD, 0, stream>>>(ffnt, tgt2, n3g, n3b, outp, nullptr, nullptr);
}
